// NodeEncoder_42477226557757
// MI455X (gfx1250) — matrix operand order repaired and checked by reading, not compiled, not run on hardware
//
#include <hip/hip_runtime.h>
#include <hip/hip_bf16.h>

// ---------------- CDNA5 WMMA types ----------------
typedef __attribute__((ext_vector_type(16))) _Float16 v16h;
typedef __attribute__((ext_vector_type(8)))  _Float16 v8h;
typedef __attribute__((ext_vector_type(8)))  float    v8f;

#define DP   320   // padded D   (300 -> 320, mult of 32)
#define DP2  640   // padded 2D  (600 -> 640, mult of 32)
#define DREAL 300
#define D2REAL 600

// ============================================================
// Pack W [L][K][Nc] (f32, row-major) -> Wt [L][Np][Kp] (f16,
// transposed + zero padded) so B fragments are contiguous.
// ============================================================
__global__ void k_packWT(const float* __restrict__ W, _Float16* __restrict__ Wt,
                         int L, int K, int Nc, int Kp, int Np) {
    long long t = (long long)blockIdx.x * blockDim.x + threadIdx.x;
    long long total = (long long)L * Np * Kp;
    if (t >= total) return;
    int k = (int)(t % Kp);
    int n = (int)((t / Kp) % Np);
    int l = (int)(t / ((long long)Kp * Np));
    float v = 0.0f;
    if (k < K && n < Nc) v = W[((size_t)l * K + k) * Nc + n];
    Wt[t] = (_Float16)v;
}

// ============================================================
// out[i,:] = x_emb1[x[i,0]] + x_emb2[x[i,1]]   (padded cols = 0)
// ============================================================
__global__ void k_init_embed(const int* __restrict__ x,
                             const float* __restrict__ e1,
                             const float* __restrict__ e2,
                             float* __restrict__ cur, int Nn) {
    long long t = (long long)blockIdx.x * blockDim.x + threadIdx.x;
    long long total = (long long)Nn * (DP / 4);
    if (t >= total) return;
    int i = (int)(t / (DP / 4));
    int c = (int)(t % (DP / 4)) * 4;
    float4 v; v.x = v.y = v.z = v.w = 0.0f;
    if (c < DREAL) {
        int a0 = x[2 * i], a1 = x[2 * i + 1];
        const float4 u1 = *(const float4*)(e1 + (size_t)a0 * DREAL + c);
        const float4 u2 = *(const float4*)(e2 + (size_t)a1 * DREAL + c);
        v.x = u1.x + u2.x; v.y = u1.y + u2.y; v.z = u1.z + u2.z; v.w = u1.w + u2.w;
    }
    *(float4*)(cur + (size_t)i * DP + c) = v;
}

// ============================================================
// agg[i,:] = cur[i,:] + edge_emb1[l][4] + edge_emb2[l][0]
// ============================================================
__global__ void k_agg_init(const float* __restrict__ cur, float* __restrict__ agg,
                           const float* __restrict__ es1, const float* __restrict__ es2,
                           int Nn) {
    long long t = (long long)blockIdx.x * blockDim.x + threadIdx.x;
    long long total = (long long)Nn * (DP / 4);
    if (t >= total) return;
    int i = (int)(t / (DP / 4));
    int c = (int)(t % (DP / 4)) * 4;
    float4 v = *(const float4*)(cur + (size_t)i * DP + c);
    if (c < DREAL) {
        const float4 u1 = *(const float4*)(es1 + c);
        const float4 u2 = *(const float4*)(es2 + c);
        v.x += u1.x + u2.x; v.y += u1.y + u2.y; v.z += u1.z + u2.z; v.w += u1.w + u2.w;
    }
    *(float4*)(agg + (size_t)i * DP + c) = v;
}

// ============================================================
// For each edge: agg[dst,:] += cur[src,:] + e1[attr0] + e2[attr1]
// One thread per (edge, 4 columns); atomics (L2 mostly resident).
// ============================================================
__global__ void k_edge_scatter(float* __restrict__ agg, const float* __restrict__ cur,
                               const int* __restrict__ eidx, const int* __restrict__ eattr,
                               const float* __restrict__ ee1, const float* __restrict__ ee2,
                               int E) {
    const int GROUPS = DREAL / 4;  // 75
    long long t = (long long)blockIdx.x * blockDim.x + threadIdx.x;
    long long total = (long long)E * GROUPS;
    if (t >= total) return;
    int e = (int)(t / GROUPS);
    int c = (int)(t % GROUPS) * 4;
    int src = eidx[e];
    int dst = eidx[E + e];
    int a0 = eattr[2 * e], a1 = eattr[2 * e + 1];
    const float4 xj = *(const float4*)(cur + (size_t)src * DP + c);
    const float4 u1 = *(const float4*)(ee1 + (size_t)a0 * DREAL + c);
    const float4 u2 = *(const float4*)(ee2 + (size_t)a1 * DREAL + c);
    float* p = agg + (size_t)dst * DP + c;
    atomicAdd(p + 0, xj.x + u1.x + u2.x);
    atomicAdd(p + 1, xj.y + u1.y + u2.y);
    atomicAdd(p + 2, xj.z + u1.z + u2.z);
    atomicAdd(p + 3, xj.w + u1.w + u2.w);
}

// ============================================================
// WMMA GEMM:  O = act(A @ Bt^T + bias)
//   A  : [M, K]  f32 (AF32=true, converted inline) or f16
//   Bt : [Ncols, K] f16 (transposed, padded)  -- B fragment is
//        two 16B v8h per lane at k0+8h and k0+16+8h (lane=column)
//   4 waves/block; each wave computes a 32x64 strip: 2 A
//   fragments x 4 B fragments = 8 v_wmma_f32_16x16x32_f16 per
//   K-step. Load rows are CLAMPED (always legal), inner loop has
//   no EXEC divergence; epilogue has an unguarded fast path for
//   interior tiles (only the last row-tile takes guarded stores).
// ============================================================
template<bool AF32, bool OUTF16, bool RELU>
__global__ __launch_bounds__(128)
void k_gemm_wmma(const void* __restrict__ Av, const _Float16* __restrict__ Bt,
                 const float* __restrict__ bias, void* __restrict__ Ov,
                 int M, int K, int lda, int ldc, int nbias) {
    const int lane = threadIdx.x & 31;
    const int wave = threadIdx.x >> 5;
    const int rows0 = blockIdx.x * 128 + wave * 32;   // wave owns 32 rows
    const int n0    = blockIdx.y * 64;                // 64 cols

    v8f zero = {};
    v8f acc[2][4];
    #pragma unroll
    for (int g = 0; g < 2; ++g)
        #pragma unroll
        for (int j = 0; j < 4; ++j) acc[g][j] = zero;

    const int asel = (lane >> 4);                     // 0: lanes 0-15, 1: lanes 16-31
    const int mlast = M - 1;
    // clamped load rows for the two 16-row groups
    int mrow[2];
    mrow[0] = rows0 + (lane & 15);       if (mrow[0] > mlast) mrow[0] = mlast;
    mrow[1] = rows0 + 16 + (lane & 15);  if (mrow[1] > mlast) mrow[1] = mlast;

    for (int k0 = 0; k0 < K; k0 += 32) {
        // ---- two A fragments (ISA 16-bit A layout) ----
        union { v16h v; v8h h[2]; } a[2];
        const int kb = k0 + asel * 8;    // halves 0..7 -> K kb..kb+7 ; 8..15 -> kb+16..kb+23
        #pragma unroll
        for (int g = 0; g < 2; ++g) {
            if (AF32) {
                const float* ap = (const float*)Av + (size_t)mrow[g] * lda + kb;
                const float4 t0 = *(const float4*)(ap + 0);
                const float4 t1 = *(const float4*)(ap + 4);
                const float4 t2 = *(const float4*)(ap + 16);
                const float4 t3 = *(const float4*)(ap + 20);
                a[g].v[0]  = (_Float16)t0.x; a[g].v[1]  = (_Float16)t0.y;
                a[g].v[2]  = (_Float16)t0.z; a[g].v[3]  = (_Float16)t0.w;
                a[g].v[4]  = (_Float16)t1.x; a[g].v[5]  = (_Float16)t1.y;
                a[g].v[6]  = (_Float16)t1.z; a[g].v[7]  = (_Float16)t1.w;
                a[g].v[8]  = (_Float16)t2.x; a[g].v[9]  = (_Float16)t2.y;
                a[g].v[10] = (_Float16)t2.z; a[g].v[11] = (_Float16)t2.w;
                a[g].v[12] = (_Float16)t3.x; a[g].v[13] = (_Float16)t3.y;
                a[g].v[14] = (_Float16)t3.z; a[g].v[15] = (_Float16)t3.w;
            } else {
                const _Float16* ap = (const _Float16*)Av + (size_t)mrow[g] * lda + kb;
                a[g].h[0] = *(const v8h*)(ap);
                a[g].h[1] = *(const v8h*)(ap + 16);
            }
        }
        // ---- 4 B fragments, each reused by both row groups ----
        const int kb2 = k0 + (asel << 3);  // halves 0..7 -> K kb2..kb2+7 ; 8..15 -> kb2+16..kb2+23
        #pragma unroll
        for (int j = 0; j < 4; ++j) {
            const int col = n0 + j * 16 + (lane & 15);
            const v8h b0 = *(const v8h*)(Bt + (size_t)col * K + kb2);
            const v8h b1 = *(const v8h*)(Bt + (size_t)col * K + kb2 + 16);
            const v16h b = __builtin_shufflevector(b0, b1, 0, 1, 2, 3, 4, 5, 6, 7, 8, 9, 10, 11, 12, 13, 14, 15);
            acc[0][j] = __builtin_amdgcn_wmma_f32_16x16x32_f16(
                false, a[0].v, false, b, (short)0, acc[0][j], false, false);
            acc[1][j] = __builtin_amdgcn_wmma_f32_16x16x32_f16(
                false, a[1].v, false, b, (short)0, acc[1][j], false, false);
        }
    }

    // ---- epilogue: bias (+ReLU), store f16 or f32 ----
    // Hoist bias (per-lane columns are fixed).
    float bv[4];
    #pragma unroll
    for (int j = 0; j < 4; ++j) {
        const int col = n0 + j * 16 + (lane & 15);
        bv[j] = (col < nbias) ? bias[col] : 0.0f;
    }

    if (rows0 + 32 <= M) {
        // Fast path: whole 32-row strip in range -> straight-line stores.
        #pragma unroll
        for (int g = 0; g < 2; ++g) {
            const int rbase = rows0 + g * 16 + (asel << 3);  // lanes16-31 hold M=8..15
            #pragma unroll
            for (int j = 0; j < 4; ++j) {
                const int col = n0 + j * 16 + (lane & 15);
                #pragma unroll
                for (int r = 0; r < 8; ++r) {
                    float v = acc[g][j][r] + bv[j];
                    if (RELU) v = fmaxf(v, 0.0f);
                    const size_t idx = (size_t)(rbase + r) * ldc + col;
                    if (OUTF16) ((_Float16*)Ov)[idx] = (_Float16)v;
                    else        ((float*)   Ov)[idx] = v;
                }
            }
        }
    } else {
        // Boundary tile: per-row guard.
        #pragma unroll
        for (int g = 0; g < 2; ++g) {
            const int rbase = rows0 + g * 16 + (asel << 3);
            #pragma unroll
            for (int j = 0; j < 4; ++j) {
                const int col = n0 + j * 16 + (lane & 15);
                #pragma unroll
                for (int r = 0; r < 8; ++r) {
                    const int row = rbase + r;
                    if (row < M) {
                        float v = acc[g][j][r] + bv[j];
                        if (RELU) v = fmaxf(v, 0.0f);
                        const size_t idx = (size_t)row * ldc + col;
                        if (OUTF16) ((_Float16*)Ov)[idx] = (_Float16)v;
                        else        ((float*)   Ov)[idx] = v;
                    }
                }
            }
        }
    }
}

// ============================================================
// BatchNorm helpers
// ============================================================
__global__ void k_zero_stats(float* __restrict__ stats) {
    int t = blockIdx.x * blockDim.x + threadIdx.x;
    if (t < 2 * DP) stats[t] = 0.0f;
}

__global__ void k_col_stats(const float* __restrict__ h, float* __restrict__ stats, int Nn) {
    int c = blockIdx.x * blockDim.x + threadIdx.x;   // < 320, coalesced across lanes
    float s = 0.0f, s2 = 0.0f;
    for (int r = blockIdx.y; r < Nn; r += gridDim.y) {
        float v = h[(size_t)r * DP + c];
        s += v; s2 += v * v;
    }
    atomicAdd(&stats[c], s);
    atomicAdd(&stats[DP + c], s2);
}

__global__ void k_bn_norm(const float* __restrict__ h, const float* __restrict__ stats,
                          const float* __restrict__ gamma, const float* __restrict__ beta,
                          float* __restrict__ out, int out_stride, int out_cols,
                          int do_relu, int Nn) {
    long long t = (long long)blockIdx.x * blockDim.x + threadIdx.x;
    long long total = (long long)Nn * out_cols;
    if (t >= total) return;
    int c = (int)(t % out_cols);
    int r = (int)(t / out_cols);
    const float invN = 1.0f / (float)Nn;
    float mean = stats[c] * invN;
    float var  = stats[DP + c] * invN - mean * mean;
    float rs   = rsqrtf(var + 1e-5f);
    float g = (c < DREAL) ? gamma[c] : 0.0f;
    float b = (c < DREAL) ? beta[c]  : 0.0f;
    float v = (h[(size_t)r * DP + c] - mean) * rs * g + b;
    if (do_relu) v = fmaxf(v, 0.0f);
    out[(size_t)r * out_stride + c] = v;
}

// ============================================================
// Host orchestration
// ============================================================
extern "C" void kernel_launch(void* const* d_in, const int* in_sizes, int n_in,
                              void* d_out, int out_size, void* d_ws, size_t ws_size,
                              hipStream_t stream) {
    const int Nn = in_sizes[0] / 2;        // nodes
    const int E  = in_sizes[1] / 2;        // edges
    const int L  = 5;

    const int*   x      = (const int*)  d_in[0];
    const int*   eidx   = (const int*)  d_in[1];
    const int*   eattr  = (const int*)  d_in[2];
    const float* x_emb1 = (const float*)d_in[3];
    const float* x_emb2 = (const float*)d_in[4];
    const float* eemb1  = (const float*)d_in[5];   // [L][6][300]
    const float* eemb2  = (const float*)d_in[6];   // [L][3][300]
    const float* W1     = (const float*)d_in[7];   // [L][300][600]
    const float* b1     = (const float*)d_in[8];   // [L][600]
    const float* W2     = (const float*)d_in[9];   // [L][600][300]
    const float* b2     = (const float*)d_in[10];  // [L][300]
    const float* gamma  = (const float*)d_in[11];  // [L][300]
    const float* beta   = (const float*)d_in[12];  // [L][300]

    // ---- workspace carve ----
    float*     cur   = (float*)d_ws;                            // [N][320] f32
    float*     agg   = cur + (size_t)Nn * DP;                   // [N][320] f32
    _Float16*  H1    = (_Float16*)(agg + (size_t)Nn * DP);      // [N][640] f16
    _Float16*  W1t   = H1  + (size_t)Nn * DP2;                  // [L][640][320] f16
    _Float16*  W2t   = W1t + (size_t)L * DP2 * DP;              // [L][320][640] f16
    float*     stats = (float*)(W2t + (size_t)L * DP * DP2);    // [640] f32

    // ---- weight pack (re-done each call: deterministic) ----
    {
        long long tot = (long long)L * DP2 * DP;
        int blk = (int)((tot + 255) / 256);
        k_packWT<<<blk, 256, 0, stream>>>(W1, W1t, L, DREAL, D2REAL, DP, DP2);
        k_packWT<<<blk, 256, 0, stream>>>(W2, W2t, L, D2REAL, DREAL, DP2, DP);
    }

    // ---- node embedding init ----
    {
        long long tot = (long long)Nn * (DP / 4);
        k_init_embed<<<(int)((tot + 255) / 256), 256, 0, stream>>>(x, x_emb1, x_emb2, cur, Nn);
    }

    for (int l = 0; l < L; ++l) {
        const float* ee1 = eemb1 + (size_t)l * 6 * DREAL;
        const float* ee2 = eemb2 + (size_t)l * 3 * DREAL;
        // agg = cur + selfloop emb (bond_type=4, dir=0)
        {
            long long tot = (long long)Nn * (DP / 4);
            k_agg_init<<<(int)((tot + 255) / 256), 256, 0, stream>>>(
                cur, agg, ee1 + (size_t)4 * DREAL, ee2, Nn);
        }
        // scatter messages
        {
            long long tot = (long long)E * (DREAL / 4);
            k_edge_scatter<<<(int)((tot + 255) / 256), 256, 0, stream>>>(
                agg, cur, eidx, eattr, ee1, ee2, E);
        }
        // GEMM1: H1 = relu(agg @ W1 + b1)   [N,320] x [320,640] -> f16 [N,640]
        {
            dim3 grid((Nn + 127) / 128, DP2 / 64);
            k_gemm_wmma<true, true, true><<<grid, 128, 0, stream>>>(
                agg, W1t + (size_t)l * DP2 * DP, b1 + (size_t)l * D2REAL,
                H1, Nn, DP, DP, DP2, D2REAL);
        }
        // GEMM2: h = H1 @ W2 + b2           [N,640] x [640,320] -> f32 [N,320] (into cur)
        {
            dim3 grid((Nn + 127) / 128, DP / 64);
            k_gemm_wmma<false, false, false><<<grid, 128, 0, stream>>>(
                H1, W2t + (size_t)l * DP * DP2, b2 + (size_t)l * DREAL,
                cur, Nn, DP2, DP2, DP, DREAL);
        }
        // BatchNorm (training stats, biased var) + ReLU (except last)
        k_zero_stats<<<3, 256, 0, stream>>>(stats);
        {
            dim3 grid(DP / 64, 512);
            k_col_stats<<<grid, 64, 0, stream>>>(cur, stats, Nn);
        }
        const bool last = (l == L - 1);
        float* outp   = last ? (float*)d_out : cur;
        int out_cols  = last ? DREAL : DP;
        long long tot = (long long)Nn * out_cols;
        k_bn_norm<<<(int)((tot + 255) / 256), 256, 0, stream>>>(
            cur, stats, gamma + (size_t)l * DREAL, beta + (size_t)l * DREAL,
            outp, out_cols, out_cols, last ? 0 : 1, Nn);
    }
}
